// LlamaAttention_33191507263918
// MI455X (gfx1250) — compile-verified, matrix operand order checked by reading, not run on hardware
//
#include <hip/hip_runtime.h>
#include <hip/hip_bf16.h>

typedef __attribute__((ext_vector_type(16))) __bf16 v16bf;
typedef __attribute__((ext_vector_type(8)))  __bf16 v8bf;
typedef __attribute__((ext_vector_type(8)))  float  v8f;

#define B_  2
#define S_  2048
#define H_  2048
#define NH_ 16
#define HD_ 128
#define NEGINF_MASK (-1.0e9f)

union BF16x16 { v16bf v; v8bf h[2]; };

static __device__ inline v16bf load_frag(const __bf16* p0, const __bf16* p1) {
    BF16x16 u;
    u.h[0] = *(const v8bf*)p0;
    u.h[1] = *(const v8bf*)p1;
    return u.v;
}

static __device__ inline v8f wmma_bf16(v16bf a, v16bf b, v8f c) {
    // D = A(16x32 bf16) x B(32x16 bf16) + C(16x16 f32)
    return __builtin_amdgcn_wmma_f32_16x16x32_bf16(
        false, a, false, b, (short)0, c, false, false);
}

static __device__ inline v8f vzero8() {
    v8f z;
#pragma unroll
    for (int i = 0; i < 8; ++i) z[i] = 0.0f;
    return z;
}

// async 16B copy global -> LDS (ASYNCcnt path, no VGPR round-trip).
// LDS byte offset == low 32 bits of the generic pointer to a __shared__ object.
static __device__ inline void async_b128(void* lds, const void* g) {
    unsigned       l  = (unsigned)(unsigned long long)(uintptr_t)lds;
    unsigned long long ga = (unsigned long long)(uintptr_t)g;
    asm volatile("global_load_async_to_lds_b128 %0, %1, off" :: "v"(l), "v"(ga) : "memory");
}
static __device__ inline void wait_async0() {
    asm volatile("s_wait_asynccnt 0x0" ::: "memory");
}

// ---------------------------------------------------------------------------
// fp32 -> bf16 cast
// ---------------------------------------------------------------------------
__global__ __launch_bounds__(256) void cast_f32_bf16(const float* __restrict__ src,
                                                     __bf16* __restrict__ dst, int n) {
    int i = blockIdx.x * blockDim.x + threadIdx.x;
    if (i < n) dst[i] = (__bf16)src[i];
}

// ---------------------------------------------------------------------------
// NT GEMM: C[M,N] = A[M,K] * W[N,K]^T   (einsum 'mk,nk->mn')
// MODE 0: fp32 row-major out.  MODE 1: bf16 out scattered to [B,NH,S,HD].
// 256 thr (8 waves), tile 128x128, wave tile 64(M)x32(N), K step 32,
// double-buffered LDS filled with async global->LDS copies.
// ---------------------------------------------------------------------------
template <int MODE>
__global__ __launch_bounds__(256) void gemm_nt(const __bf16* __restrict__ A,
                                               const __bf16* __restrict__ W,
                                               void* __restrict__ out,
                                               int M, int N, int K) {
    __shared__ __bf16 As[2][128 * 32];
    __shared__ __bf16 Bs[2][128 * 32];

    const int t    = threadIdx.x;
    const int lane = t & 31;
    const int wave = t >> 5;
    const int wm   = wave >> 2;      // 0..1 -> M offset wm*64
    const int wn   = wave & 3;       // 0..3 -> N offset wn*32
    const int half = lane >> 4;      // K-half selector per ISA 16-bit layout
    const int r    = lane & 15;

    const __bf16* Arow = A + (size_t)(blockIdx.y * 128) * K;
    const __bf16* Brow = W + (size_t)(blockIdx.x * 128) * K;

    v8f acc[4][2];
#pragma unroll
    for (int i = 0; i < 4; ++i)
#pragma unroll
        for (int j = 0; j < 2; ++j) acc[i][j] = vzero8();

    // issue async fill of buffer 0 (4 x b128 per thread = full 128x32 A+B tiles)
#pragma unroll
    for (int it = 0; it < 2; ++it) {
        int idx = t + it * 256, row = idx >> 2, ck = (idx & 3) * 8;
        async_b128(&As[0][row * 32 + ck], Arow + (size_t)row * K + ck);
        async_b128(&Bs[0][row * 32 + ck], Brow + (size_t)row * K + ck);
    }

    int buf = 0;
    for (int k0 = 0; k0 < K; k0 += 32) {
        wait_async0();           // my previous fill landed in LDS
        __syncthreads();         // everyone's fill landed; prior compute done
        if (k0 + 32 < K) {       // prefetch next slice into the other buffer
#pragma unroll
            for (int it = 0; it < 2; ++it) {
                int idx = t + it * 256, row = idx >> 2, ck = (idx & 3) * 8;
                async_b128(&As[buf ^ 1][row * 32 + ck], Arow + (size_t)row * K + k0 + 32 + ck);
                async_b128(&Bs[buf ^ 1][row * 32 + ck], Brow + (size_t)row * K + k0 + 32 + ck);
            }
        }

        const __bf16* as = As[buf];
        const __bf16* bs = Bs[buf];
        v16bf bfr[2];
#pragma unroll
        for (int j = 0; j < 2; ++j) {
            int rowb = wn * 32 + j * 16 + r;
            bfr[j] = load_frag(bs + rowb * 32 + half * 8,
                               bs + rowb * 32 + 16 + half * 8);
        }
#pragma unroll
        for (int i = 0; i < 4; ++i) {
            int rowa = wm * 64 + i * 16 + r;
            v16bf af = load_frag(as + rowa * 32 + half * 8,
                                 as + rowa * 32 + 16 + half * 8);
#pragma unroll
            for (int j = 0; j < 2; ++j) acc[i][j] = wmma_bf16(af, bfr[j], acc[i][j]);
        }
        buf ^= 1;
    }

    // epilogue: C layout = lanes0-15 row e, lanes16-31 row e+8; col = lane&15
    const int mbase  = blockIdx.y * 128 + wm * 64;
    const int nbase  = blockIdx.x * 128 + wn * 32;
    const int rowoff = (lane >= 16) ? 8 : 0;
    const int col    = lane & 15;
#pragma unroll
    for (int i = 0; i < 4; ++i)
#pragma unroll
        for (int j = 0; j < 2; ++j)
#pragma unroll
            for (int e = 0; e < 8; ++e) {
                int m = mbase + i * 16 + e + rowoff;
                int n = nbase + j * 16 + col;
                float val = acc[i][j][e];
                if (MODE == 0) {
                    ((float*)out)[(size_t)m * N + n] = val;
                } else {
                    int b = m >> 11, s = m & (S_ - 1);
                    int head = n >> 7, d = n & (HD_ - 1);
                    ((__bf16*)out)[((((size_t)b * NH_ + head) * S_) + s) * HD_ + d] = (__bf16)val;
                }
            }
}

// ---------------------------------------------------------------------------
// RoPE in place on bf16 Q and K, layout [B*NH, S, HD]
// ---------------------------------------------------------------------------
__global__ __launch_bounds__(256) void rope_kernel(__bf16* __restrict__ Qb,
                                                   __bf16* __restrict__ Kb) {
    int tid = blockIdx.x * blockDim.x + threadIdx.x;   // 2*32*2048*64 threads
    int i      = tid & 63;
    int s      = (tid >> 6) & (S_ - 1);
    int bh     = (tid >> 17) & 31;
    int tensor = tid >> 22;
    __bf16* p  = (tensor ? Kb : Qb) + ((size_t)bh * S_ + s) * HD_;

    float invf  = __expf(-(float)i * (9.2103403719761836f / 64.0f)); // 10000^(-i/64)
    float ang   = (float)s * invf;
    float sn, cs;
    __sincosf(ang, &sn, &cs);
    float x1 = (float)p[i];
    float x2 = (float)p[i + 64];
    p[i]      = (__bf16)(x1 * cs - x2 * sn);
    p[i + 64] = (__bf16)(x2 * cs + x1 * sn);
}

// ---------------------------------------------------------------------------
// Flash attention: grid (B*NH, S/64), block 128 thr = 4 waves, 16 q-rows/wave.
// K tile staged async into LDS (shared by all 4 waves); V tile transposed
// into LDS; P transposed through per-wave LDS. All matmuls bf16 WMMA.
// ---------------------------------------------------------------------------
__global__ __launch_bounds__(128) void attn_kernel(const __bf16* __restrict__ Qb,
                                                   const __bf16* __restrict__ Kb,
                                                   const __bf16* __restrict__ Vb,
                                                   __bf16* __restrict__ AOb) {
    __shared__ __bf16 Ks[64 * HD_];        // [k][d] natural K tile (16 KB)
    __shared__ __bf16 Vt[HD_ * 64];        // [d][k] transposed V tile (16 KB)
    __shared__ __bf16 Pbuf[4 * 16 * 64];   // per-wave P tile (8 KB)

    const int t    = threadIdx.x;
    const int lane = t & 31;
    const int w    = t >> 5;
    const int half = lane >> 4;
    const int r    = lane & 15;
    const int rowoff = half ? 8 : 0;

    const int bh = blockIdx.x;
    const int q0 = blockIdx.y * 64;
    const size_t base = (size_t)bh * S_ * HD_;
    const float scale = 0.088388347648318447f;   // 1/sqrt(128)

    // preload Q fragments (16 rows x 128 d = 4 K-chunks of 32)
    v16bf qf[4];
    {
        int qrow = q0 + w * 16 + r;
        const __bf16* qp = Qb + base + (size_t)qrow * HD_;
#pragma unroll
        for (int c = 0; c < 4; ++c)
            qf[c] = load_frag(qp + c * 32 + half * 8, qp + c * 32 + 16 + half * 8);
    }

    float m[8], l[8];
    v8f O[8];
#pragma unroll
    for (int e = 0; e < 8; ++e) { m[e] = -3.0e38f; l[e] = 0.0f; }
#pragma unroll
    for (int d = 0; d < 8; ++d) O[d] = vzero8();

    for (int j0 = 0; j0 <= q0; j0 += 64) {       // causal: skip blocks past diagonal
        // stage K tile async into LDS (64 rows x 128 d, natural layout)
#pragma unroll
        for (int it = 0; it < 8; ++it) {
            int cid = t + it * 128;              // 1024 chunks of 8 bf16
            int kv  = cid >> 4;
            int dc  = (cid & 15) * 8;
            async_b128(&Ks[kv * HD_ + dc], Kb + base + (size_t)(j0 + kv) * HD_ + dc);
        }
        // stage V tile transposed into LDS: Vt[d][k]
#pragma unroll
        for (int it = 0; it < 8; ++it) {
            int cid = t + it * 128;
            int kv  = cid >> 4;
            int dc  = (cid & 15) * 8;
            v8bf vv = *(const v8bf*)(Vb + base + (size_t)(j0 + kv) * HD_ + dc);
#pragma unroll
            for (int x = 0; x < 8; ++x) Vt[(dc + x) * 64 + kv] = vv[x];
        }
        wait_async0();
        __syncthreads();

        // S = Q K^T  (16 x 64 per wave, 4 n-tiles x 4 k-chunks) from LDS K
        v8f sc[4];
#pragma unroll
        for (int ti = 0; ti < 4; ++ti) {
            sc[ti] = vzero8();
            const __bf16* kp = Ks + (ti * 16 + r) * HD_;
#pragma unroll
            for (int c = 0; c < 4; ++c) {
                v16bf kf = load_frag(kp + c * 32 + half * 8, kp + c * 32 + 16 + half * 8);
                sc[ti] = wmma_bf16(qf[c], kf, sc[ti]);
            }
        }

        // scale + causal mask
#pragma unroll
        for (int ti = 0; ti < 4; ++ti)
#pragma unroll
            for (int e = 0; e < 8; ++e) {
                int kcol = j0 + ti * 16 + r;
                int qrow = q0 + w * 16 + e + rowoff;
                float s = sc[ti][e] * scale;
                if (kcol > qrow) s += NEGINF_MASK;
                sc[ti][e] = s;
            }

        // online softmax: row reductions across 16-lane halves
        float mnew[8], alpha[8];
#pragma unroll
        for (int e = 0; e < 8; ++e) {
            float v = sc[0][e];
#pragma unroll
            for (int ti = 1; ti < 4; ++ti) v = fmaxf(v, sc[ti][e]);
#pragma unroll
            for (int msk = 1; msk < 16; msk <<= 1)
                v = fmaxf(v, __shfl_xor(v, msk, 32));
            mnew[e]  = fmaxf(m[e], v);
            alpha[e] = __expf(m[e] - mnew[e]);
            m[e]     = mnew[e];
        }
#pragma unroll
        for (int ti = 0; ti < 4; ++ti)
#pragma unroll
            for (int e = 0; e < 8; ++e)
                sc[ti][e] = __expf(sc[ti][e] - mnew[e]);
#pragma unroll
        for (int e = 0; e < 8; ++e) {
            float rs = sc[0][e] + sc[1][e] + sc[2][e] + sc[3][e];
#pragma unroll
            for (int msk = 1; msk < 16; msk <<= 1)
                rs += __shfl_xor(rs, msk, 32);
            l[e] = l[e] * alpha[e] + rs;
        }
#pragma unroll
        for (int d = 0; d < 8; ++d)
#pragma unroll
            for (int e = 0; e < 8; ++e) O[d][e] *= alpha[e];

        // transpose P through per-wave LDS (C layout -> A layout)
        __bf16* Pw = Pbuf + w * 16 * 64;
#pragma unroll
        for (int ti = 0; ti < 4; ++ti)
#pragma unroll
            for (int e = 0; e < 8; ++e)
                Pw[(e + rowoff) * 64 + ti * 16 + r] = (__bf16)sc[ti][e];
        asm volatile("s_wait_dscnt 0x0" ::: "memory");  // wave-local LDS RAW

        // O += P V  (2 K-chunks of 32 over the 64 keys, 8 d-tiles)
#pragma unroll
        for (int c2 = 0; c2 < 2; ++c2) {
            v16bf pf = load_frag(Pw + r * 64 + c2 * 32 + half * 8,
                                 Pw + r * 64 + c2 * 32 + 16 + half * 8);
#pragma unroll
            for (int dt = 0; dt < 8; ++dt) {
                v16bf vf = load_frag(Vt + (dt * 16 + r) * 64 + c2 * 32 + half * 8,
                                     Vt + (dt * 16 + r) * 64 + c2 * 32 + 16 + half * 8);
                O[dt] = wmma_bf16(pf, vf, O[dt]);
            }
        }
        __syncthreads();
    }

    // normalize + store to row-major [B*S, H] bf16
    const int b = bh >> 4, head = bh & 15;
#pragma unroll
    for (int dt = 0; dt < 8; ++dt)
#pragma unroll
        for (int e = 0; e < 8; ++e) {
            int q = q0 + w * 16 + e + rowoff;
            size_t mrow = (size_t)b * S_ + q;
            int n = head * HD_ + dt * 16 + r;
            AOb[mrow * H_ + n] = (__bf16)(O[dt][e] / l[e]);
        }
}

// ---------------------------------------------------------------------------
extern "C" void kernel_launch(void* const* d_in, const int* in_sizes, int n_in,
                              void* d_out, int out_size, void* d_ws, size_t ws_size,
                              hipStream_t stream) {
    const float* hidden = (const float*)d_in[0];
    // d_in[1] = attention_mask (causal; computed analytically in-kernel)
    const float* Wq = (const float*)d_in[2];
    const float* Wk = (const float*)d_in[3];
    const float* Wv = (const float*)d_in[4];
    const float* Wo = (const float*)d_in[5];

    const int M = B_ * S_;           // 4096
    const int N = H_;                // 2048
    const int K = H_;                // 2048
    const int nHid = M * H_;         // 8,388,608
    const int nW   = H_ * H_;        // 4,194,304
    const size_t nQKV = (size_t)B_ * NH_ * S_ * HD_;  // 8,388,608

    char* w = (char*)d_ws;
    __bf16* Xb  = (__bf16*)w; w += (size_t)nHid * 2;
    __bf16* Wqb = (__bf16*)w; w += (size_t)nW * 2;
    __bf16* Wkb = (__bf16*)w; w += (size_t)nW * 2;
    __bf16* Wvb = (__bf16*)w; w += (size_t)nW * 2;
    __bf16* Wob = (__bf16*)w; w += (size_t)nW * 2;
    __bf16* Qb  = (__bf16*)w; w += nQKV * 2;
    __bf16* Kb  = (__bf16*)w; w += nQKV * 2;
    __bf16* Vb  = (__bf16*)w; w += nQKV * 2;
    __bf16* AOb = (__bf16*)w; w += (size_t)nHid * 2;

    cast_f32_bf16<<<(nHid + 255) / 256, 256, 0, stream>>>(hidden, Xb, nHid);
    cast_f32_bf16<<<(nW + 255) / 256, 256, 0, stream>>>(Wq, Wqb, nW);
    cast_f32_bf16<<<(nW + 255) / 256, 256, 0, stream>>>(Wk, Wkb, nW);
    cast_f32_bf16<<<(nW + 255) / 256, 256, 0, stream>>>(Wv, Wvb, nW);
    cast_f32_bf16<<<(nW + 255) / 256, 256, 0, stream>>>(Wo, Wob, nW);

    dim3 ggrid(N / 128, M / 128);    // (16, 32)
    gemm_nt<1><<<ggrid, 256, 0, stream>>>(Xb, Wqb, Qb, M, N, K);
    gemm_nt<1><<<ggrid, 256, 0, stream>>>(Xb, Wkb, Kb, M, N, K);
    gemm_nt<1><<<ggrid, 256, 0, stream>>>(Xb, Wvb, Vb, M, N, K);

    // 2 tensors * 32 (b,h) * 2048 s * 64 pairs = 8,388,608 threads
    rope_kernel<<<8388608 / 256, 256, 0, stream>>>(Qb, Kb);

    attn_kernel<<<dim3(B_ * NH_, S_ / 64), 128, 0, stream>>>(Qb, Kb, Vb, AOb);

    gemm_nt<0><<<ggrid, 256, 0, stream>>>(AOb, Wob, d_out, M, N, K);
}
